// TextImageMatchingLoss_87479893885712
// MI455X (gfx1250) — matrix operand order repaired and checked by reading, not compiled, not run on hardware
//
#include <hip/hip_runtime.h>
#include <hip/hip_bf16.h>

#define BN 8192
#define DK 512
#define MARGIN 0.2f
#define GEMM_BLOCKS 2048   // 16384 wave-tiles (128x128 tiles of 64x64) / 8 waves per block

typedef __attribute__((ext_vector_type(16))) _Float16 v16h;
typedef __attribute__((ext_vector_type(8)))  _Float16 v8h;
typedef __attribute__((ext_vector_type(8)))  float    v8f;

struct __align__(8) h4 { _Float16 a, b, c, d; };

// ---------------------------------------------------------------------------
// Kernel 1: L2-normalize rows of img & txt into fp16, compute pos[i] =
// <img_n[i], txt_n[i]>. One wave (32 lanes) per row; each lane owns 4 float4s.
// ---------------------------------------------------------------------------
__global__ __launch_bounds__(256) void nrm_kernel(
    const float* __restrict__ img, const float* __restrict__ txt,
    _Float16* __restrict__ imgH, _Float16* __restrict__ txtH,
    float* __restrict__ pos)
{
    const int wave = threadIdx.x >> 5;
    const int lane = threadIdx.x & 31;
    const int row  = blockIdx.x * 8 + wave;

    const float4* ip = (const float4*)(img + (size_t)row * DK);
    const float4* tp = (const float4*)(txt + (size_t)row * DK);

    float4 iv[4], tv[4];
    float si = 0.f, st = 0.f, sx = 0.f;
#pragma unroll
    for (int t = 0; t < 4; ++t) {
        iv[t] = ip[lane + 32 * t];
        tv[t] = tp[lane + 32 * t];
        si += iv[t].x * iv[t].x + iv[t].y * iv[t].y + iv[t].z * iv[t].z + iv[t].w * iv[t].w;
        st += tv[t].x * tv[t].x + tv[t].y * tv[t].y + tv[t].z * tv[t].z + tv[t].w * tv[t].w;
        sx += iv[t].x * tv[t].x + iv[t].y * tv[t].y + iv[t].z * tv[t].z + iv[t].w * tv[t].w;
    }
    // wave32 butterfly reduction
#pragma unroll
    for (int o = 16; o > 0; o >>= 1) {
        si += __shfl_xor(si, o, 32);
        st += __shfl_xor(st, o, 32);
        sx += __shfl_xor(sx, o, 32);
    }
    const float nI = fmaxf(sqrtf(si), 1e-12f);
    const float nT = fmaxf(sqrtf(st), 1e-12f);
    const float rI = 1.0f / nI;
    const float rT = 1.0f / nT;
    if (lane == 0) pos[row] = sx * rI * rT;

    h4* oi = (h4*)(imgH + (size_t)row * DK);
    h4* ot = (h4*)(txtH + (size_t)row * DK);
#pragma unroll
    for (int t = 0; t < 4; ++t) {
        h4 a, b;
        a.a = (_Float16)(iv[t].x * rI); a.b = (_Float16)(iv[t].y * rI);
        a.c = (_Float16)(iv[t].z * rI); a.d = (_Float16)(iv[t].w * rI);
        b.a = (_Float16)(tv[t].x * rT); b.b = (_Float16)(tv[t].y * rT);
        b.c = (_Float16)(tv[t].z * rT); b.d = (_Float16)(tv[t].w * rT);
        oi[lane + 32 * t] = a;
        ot[lane + 32 * t] = b;
    }
}

// ---------------------------------------------------------------------------
// Kernel 2: fused GEMM (v_wmma_f32_16x16x32_f16) + margin-relu epilogue.
// Each wave computes a 64x64 tile of sim = imgH @ txtH^T (4x4 WMMA tiles:
// 16 v_wmma per K-step fed by 16 b128 loads -> 32 FLOP/B from L2), then
// folds relu(m - pos[i] + s) + relu(m - pos[j] + s) over i != j and reduces.
// The i==j exclusion is only possible when tI == tJ (scalar-uniform), so
// the common path is a completely unguarded epilogue.
// ---------------------------------------------------------------------------
__global__ __launch_bounds__(256) void gemm_loss_kernel(
    const _Float16* __restrict__ imgH, const _Float16* __restrict__ txtH,
    const float* __restrict__ pos, float* __restrict__ partial)
{
    const int wave = threadIdx.x >> 5;
    const int lane = threadIdx.x & 31;
    const int tile = blockIdx.x * 8 + wave;      // 0 .. 16383
    const int tI = tile >> 7;                    // 0 .. 127
    const int tJ = tile & 127;
    const int I0 = tI * 64;
    const int J0 = tJ * 64;
    const int lr = lane & 15;
    const int hi = lane >> 4;

    // WMMA fragment layouts used here (contiguous K, same order for A and B):
    //  A (16-bit, 16x32): lanes 0-15 = row lr, K 0..15;
    //                     lanes 16-31 = row lr, K 16..31.
    //  B (16-bit, 32x16): lane = col; lanes 0-15 hold K 0..15, 16-31 hold 16..31.
    const _Float16* aBase = imgH + (size_t)(I0 + lr) * DK + hi * 16;
    const _Float16* bBase = txtH + (size_t)(J0 + lr) * DK + hi * 16;

    v8f acc[4][4] = {};

    for (int k0 = 0; k0 < DK; k0 += 32) {
        v16h a[4], b[4];
#pragma unroll
        for (int r = 0; r < 4; ++r) {
            const v8h* p = (const v8h*)(aBase + (size_t)r * 16 * DK + k0);
            v8h lo = p[0];   // K = k0 + hi*16 + [0..7]
            v8h hg = p[1];   // K = k0 + hi*16 + 8 + [0..7]
            v16h av;
#pragma unroll
            for (int e = 0; e < 8; ++e) { av[e] = lo[e]; av[8 + e] = hg[e]; }
            a[r] = av;
        }
#pragma unroll
        for (int c = 0; c < 4; ++c) {
            b[c] = *(const v16h*)(bBase + (size_t)c * 16 * DK + k0);  // 16 contiguous halves
        }
#pragma unroll
        for (int r = 0; r < 4; ++r)
#pragma unroll
            for (int c = 0; c < 4; ++c)
                acc[r][c] = __builtin_amdgcn_wmma_f32_16x16x32_f16(
                    false, a[r], false, b[c], (short)0, acc[r][c], false, false);
    }

    // Epilogue: C/D layout — VGPR v holds M = v + 8*hi, lane&15 = N.
    float lsum = 0.f;
    if (tI != tJ) {
        // Off-diagonal macro-tile: no i==j possible -> unguarded hot path.
#pragma unroll
        for (int r = 0; r < 4; ++r) {
            const int ibase = I0 + 16 * r + 8 * hi;
            float pi[8];
#pragma unroll
            for (int v = 0; v < 8; ++v) pi[v] = pos[ibase + v];
#pragma unroll
            for (int c = 0; c < 4; ++c) {
                const float pj = pos[J0 + 16 * c + lr];
#pragma unroll
                for (int v = 0; v < 8; ++v) {
                    const float s = acc[r][c][v];
                    lsum += fmaxf(MARGIN - pi[v] + s, 0.f)
                          + fmaxf(MARGIN - pj   + s, 0.f);
                }
            }
        }
    } else {
        // Diagonal macro-tile: branchless per-element select (v_cndmask).
#pragma unroll
        for (int r = 0; r < 4; ++r) {
            const int ibase = I0 + 16 * r + 8 * hi;
            float pi[8];
#pragma unroll
            for (int v = 0; v < 8; ++v) pi[v] = pos[ibase + v];
#pragma unroll
            for (int c = 0; c < 4; ++c) {
                const int j = J0 + 16 * c + lr;
                const float pj = pos[j];
#pragma unroll
                for (int v = 0; v < 8; ++v) {
                    const float s = acc[r][c][v];
                    const float t = fmaxf(MARGIN - pi[v] + s, 0.f)
                                  + fmaxf(MARGIN - pj   + s, 0.f);
                    lsum += (ibase + v != j) ? t : 0.f;
                }
            }
        }
    }
#pragma unroll
    for (int o = 16; o > 0; o >>= 1) lsum += __shfl_xor(lsum, o, 32);

    __shared__ float wsum[8];
    if (lane == 0) wsum[wave] = lsum;
    __syncthreads();
    if (threadIdx.x == 0) {
        float s = 0.f;
#pragma unroll
        for (int w = 0; w < 8; ++w) s += wsum[w];
        partial[blockIdx.x] = s;
    }
}

// ---------------------------------------------------------------------------
// Kernel 3: reduce block partials -> loss / (B*(B-1))
// ---------------------------------------------------------------------------
__global__ __launch_bounds__(256) void reduce_kernel(
    const float* __restrict__ partial, float* __restrict__ out)
{
    float s = 0.f;
    for (int i = threadIdx.x; i < GEMM_BLOCKS; i += 256) s += partial[i];
    __shared__ float sm[256];
    sm[threadIdx.x] = s;
    __syncthreads();
    for (int st = 128; st > 0; st >>= 1) {
        if (threadIdx.x < st) sm[threadIdx.x] += sm[threadIdx.x + st];
        __syncthreads();
    }
    if (threadIdx.x == 0)
        out[0] = sm[0] * (1.0f / ((float)BN * (float)(BN - 1)));
}

extern "C" void kernel_launch(void* const* d_in, const int* in_sizes, int n_in,
                              void* d_out, int out_size, void* d_ws, size_t ws_size,
                              hipStream_t stream) {
    (void)in_sizes; (void)n_in; (void)out_size; (void)ws_size;
    const float* img = (const float*)d_in[0];
    const float* txt = (const float*)d_in[1];

    // Workspace layout: imgH[8MB] | txtH[8MB] | pos[32KB] | partial[8KB]
    char* ws = (char*)d_ws;
    _Float16* imgH    = (_Float16*)ws;
    _Float16* txtH    = imgH + (size_t)BN * DK;
    float*    pos     = (float*)(ws + 2 * (size_t)BN * DK * sizeof(_Float16));
    float*    partial = pos + BN;

    nrm_kernel<<<BN / 8, 256, 0, stream>>>(img, txt, imgH, txtH, pos);
    gemm_loss_kernel<<<GEMM_BLOCKS, 256, 0, stream>>>(imgH, txtH, pos, partial);
    reduce_kernel<<<1, 256, 0, stream>>>(partial, (float*)d_out);
}
